// Node_Model_30777735643492
// MI455X (gfx1250) — compile-verified, matrix operand order checked by reading, not run on hardware
//
#include <hip/hip_runtime.h>
#include <hip/hip_bf16.h>

typedef __attribute__((ext_vector_type(16))) _Float16 v16h;
typedef __attribute__((ext_vector_type(8)))  _Float16 v8h;
typedef __attribute__((ext_vector_type(8)))  float    v8f;

// types for the gfx1250 async global->LDS builtin (per hipcc diagnostic:
// param is '__attribute__((__vector_size__(4 * sizeof(int)))) int __device__ *')
typedef __attribute__((vector_size(16))) int v4i_b128;
typedef __attribute__((address_space(1))) v4i_b128* g_b128ptr;
typedef __attribute__((address_space(3))) v4i_b128* l_b128ptr;

#define NNODES 100000
#define NEDGES 640000
#define HID 128

#if __has_builtin(__builtin_amdgcn_global_load_async_to_lds_b128) && \
    __has_builtin(__builtin_amdgcn_s_wait_asynccnt)
#define ASYNC_LDS 1
#else
#define ASYNC_LDS 0
#endif

// ---------------- workspace layout (bytes) ----------------
// W2p and W3p contiguous (one 64KB async stage); W4p and W5p contiguous (51KB).
#define OFF_W2P   0               // 16384 halfs = 32768 B
#define OFF_W3P   32768           // 16384 halfs = 32768 B
#define OFF_W4P   65536           // 23040 halfs = 46080 B
#define OFF_W5P   111616          // 2560 halfs  = 5120 B
#define OFF_SUMS  116736          // NNODES*128 floats
#define OFF_CNT   (116736 + (size_t)NNODES*HID*4)  // NNODES floats

// ---------------- dynamic LDS layouts (bytes) ----------------
// edge kernel
#define E_W2   0                  // 32768
#define E_W3   32768              // 32768
#define E_W1   65536              // 384 floats = 1536
#define E_B1   67072              // 128 floats
#define E_B2   67584
#define E_B3   68096
#define E_SCR  68608              // 8 waves * 4096 B (16x128 halfs each)
#define E_SMEM (68608 + 8 * 4096) // 101376 B
// node kernel
#define N_W4   0                  // 46080
#define N_W5   46080              // 5120
#define N_B4   51200              // 144 floats = 576
#define N_B5   51776              // 16 floats = 64
#define N_SCR  51840              // 8 waves * 5120 B (16x160 halfs each)
#define N_SMEM (51840 + 8 * 5120) // 92800 B

// Packed B-fragment base index: fragment (kt,nt), 16 halfs per lane.
// Element i of lane maps to K = kt*32 + (i>>3)*16 + (lane>>4)*8 + (i&7),
//                           N = nt*16 + (lane&15)     (CDNA5 16-bit B layout)
__device__ __host__ __forceinline__ int frag_idx(int kt, int nt, int NT, int lane) {
  return (((kt * NT + nt) * 32 + lane) << 4);
}

// async (or fallback) copy of `bytes` (16B multiple) global -> LDS, block-cooperative
__device__ __forceinline__ void stage_to_lds(const void* gsrc, char* smem_dst,
                                             int bytes, int tid, int nthr) {
#if ASYNC_LDS
  const char* g = (const char*)gsrc;
  for (int i = tid * 16; i < bytes; i += nthr * 16) {
    __builtin_amdgcn_global_load_async_to_lds_b128(
        (g_b128ptr)(g + i), (l_b128ptr)(smem_dst + i), 0, 0);
  }
#else
  const int4* g = (const int4*)gsrc;
  int4* d = (int4*)smem_dst;
  for (int i = tid; i < bytes / 16; i += nthr) d[i] = g[i];
#endif
}

__device__ __forceinline__ void stage_fence() {
#if ASYNC_LDS
  __builtin_amdgcn_s_wait_asynccnt(0);
#endif
  __syncthreads();
}

// ---------------- weight pre-swizzle + f16 convert ----------------
__device__ __forceinline__ void decomp(int id, int NT, int& kt, int& nt, int& K, int& N) {
  int i    = id & 15;
  int lane = (id >> 4) & 31;
  int rest = id >> 9;
  nt = rest % NT;
  kt = rest / NT;
  K = kt * 32 + ((i >> 3) << 4) + ((lane >> 4) << 3) + (i & 7);
  N = nt * 16 + (lane & 15);
}

__global__ void prep_kernel(const float* __restrict__ W2, const float* __restrict__ W3,
                            const float* __restrict__ W4, const float* __restrict__ W5,
                            _Float16* __restrict__ W2p, _Float16* __restrict__ W3p,
                            _Float16* __restrict__ W4p, _Float16* __restrict__ W5p) {
  const int S2 = 4 * 8 * 512;   // 16384
  const int S4 = 5 * 9 * 512;   // 23040
  const int S5 = 5 * 1 * 512;   // 2560
  int id = blockIdx.x * blockDim.x + threadIdx.x;
  int kt, nt, K, N;
  if (id < S2) {
    decomp(id, 8, kt, nt, K, N);
    W2p[id] = (_Float16)W2[K * 128 + N];
  } else if (id < 2 * S2) {
    int j = id - S2;
    decomp(j, 8, kt, nt, K, N);
    W3p[j] = (_Float16)W3[K * 128 + N];
  } else if (id < 2 * S2 + S4) {
    int j = id - 2 * S2;
    decomp(j, 9, kt, nt, K, N);                       // K padded to 160, N to 144
    W4p[j] = (K < 131 && N < 131) ? (_Float16)W4[K * 131 + N] : (_Float16)0.f;
  } else if (id < 2 * S2 + S4 + S5) {
    int j = id - 2 * S2 - S4;
    decomp(j, 1, kt, nt, K, N);                       // K padded to 160, N to 16
    W5p[j] = (K < 131 && N < 2) ? (_Float16)W5[K * 2 + N] : (_Float16)0.f;
  }
}

// ---------------- fused edge MLP + scatter-add ----------------
// One wave (32 lanes) processes one 16-edge tile per loop iteration.
__global__ void __launch_bounds__(256)
edge_kernel(const float* __restrict__ x, const int* __restrict__ ei,
            const float* __restrict__ ea,
            const float* __restrict__ W1, const float* __restrict__ b1,
            const _Float16* __restrict__ W2p,   // W3p = W2p + 16384 (contiguous)
            const float* __restrict__ b2, const float* __restrict__ b3,
            float* __restrict__ sums, float* __restrict__ cnt) {
  extern __shared__ char smem[];
  _Float16* sW2 = (_Float16*)(smem + E_W2);
  _Float16* sW3 = (_Float16*)(smem + E_W3);
  float*    sW1 = (float*)(smem + E_W1);
  float*    sB1 = (float*)(smem + E_B1);
  float*    sB2 = (float*)(smem + E_B2);
  float*    sB3 = (float*)(smem + E_B3);

  const int tid = threadIdx.x;
  // async-stage both weight matrices' fragments (64 KB) into LDS
  stage_to_lds(W2p, smem + E_W2, 2 * 32768, tid, blockDim.x);
  for (int i = tid; i < 384; i += blockDim.x) sW1[i] = W1[i];
  if (tid < 128) { sB1[tid] = b1[tid]; sB2[tid] = b2[tid]; sB3[tid] = b3[tid]; }
  stage_fence();

  const int lane = tid & 31;
  const int wid  = tid >> 5;
  const int hl   = lane >> 4;    // which half of the wave
  const int lm   = lane & 15;
  const int nWaves = (gridDim.x * blockDim.x) >> 5;
  const int wId    = (blockIdx.x * blockDim.x + tid) >> 5;
  const int ntiles = NEDGES / 16;
  _Float16* my = (_Float16*)(smem + E_SCR) + wid * 2048;   // 16x128 halfs

  for (int t = wId; t < ntiles; t += nWaves) {
    __builtin_prefetch(&ei[(t + nWaves) * 16], 0, 0);  // global_prefetch: next tile idx

    // lanes 0..15 gather one edge each
    int dstL = 0; float x0L = 0.f, x1L = 0.f, eaL = 0.f;
    if (lane < 16) {
      int e   = t * 16 + lane;
      int src = ei[e];
      dstL    = ei[NEDGES + e];
      x0L = x[src * 2 + 0];
      x1L = x[src * 2 + 1];
      eaL = ea[e];
    }
    const float x0m = __shfl(x0L, lm, 32);
    const float x1m = __shfl(x1L, lm, 32);
    const float eam = __shfl(eaL, lm, 32);

    // ---- layer 1 (K=3) computed straight into A-fragment layout ----
    v16h a1[4];
#pragma unroll
    for (int kt = 0; kt < 4; ++kt) {
#pragma unroll
      for (int i = 0; i < 16; ++i) {
        int K = kt * 32 + ((i >> 3) << 4) + (hl << 3) + (i & 7);
        float h = fmaf(x0m, sW1[K],
                  fmaf(x1m, sW1[128 + K],
                  fmaf(eam, sW1[256 + K], sB1[K])));
        a1[kt][i] = (_Float16)fmaxf(h, 0.f);
      }
    }

    // ---- layer 2: h2 = relu(h1 @ W2 + b2)  (32 WMMAs) ----
#pragma unroll
    for (int nt = 0; nt < 8; ++nt) {
      v8f acc = {};
#pragma unroll
      for (int kt = 0; kt < 4; ++kt) {
        v16h b = *(const v16h*)&sW2[frag_idx(kt, nt, 8, lane)];
        acc = __builtin_amdgcn_wmma_f32_16x16x32_f16(false, a1[kt], false, b,
                                                     (short)0, acc, false, false);
      }
      float bias = sB2[nt * 16 + lm];
#pragma unroll
      for (int r = 0; r < 8; ++r) {
        int M = (hl << 3) + r;
        my[M * 128 + nt * 16 + lm] = (_Float16)fmaxf(acc[r] + bias, 0.f);
      }
    }

    // reload h2 as A fragments (C-layout -> A-layout transpose via LDS)
    v16h a2[4];
#pragma unroll
    for (int kt = 0; kt < 4; ++kt) {
#pragma unroll
      for (int g = 0; g < 2; ++g) {
        int Kb = kt * 32 + g * 16 + (hl << 3);
        v8h c8 = *(const v8h*)&my[lm * 128 + Kb];
#pragma unroll
        for (int q = 0; q < 8; ++q) a2[kt][g * 8 + q] = c8[q];
      }
    }

    // ---- layer 3: h3 = h2 @ W3 + b3, scatter-add to destination nodes ----
#pragma unroll
    for (int nt = 0; nt < 8; ++nt) {
      v8f acc = {};
#pragma unroll
      for (int kt = 0; kt < 4; ++kt) {
        v16h b = *(const v16h*)&sW3[frag_idx(kt, nt, 8, lane)];
        acc = __builtin_amdgcn_wmma_f32_16x16x32_f16(false, a2[kt], false, b,
                                                     (short)0, acc, false, false);
      }
      float bias = sB3[nt * 16 + lm];
#pragma unroll
      for (int r = 0; r < 8; ++r) {
        int M    = (hl << 3) + r;          // M in [0,16)
        int node = __shfl(dstL, M, 32);
        atomicAdd(&sums[node * HID + nt * 16 + lm], acc[r] + bias);
      }
    }
    if (lane < 16) atomicAdd(&cnt[dstL], 1.0f);
  }
}

// ---------------- fused mean + node MLP ----------------
__global__ void __launch_bounds__(256)
node_kernel(const float* __restrict__ x, const float* __restrict__ u,
            const int* __restrict__ batch,
            const _Float16* __restrict__ W4p,  // W5p = W4p + 23040 (contiguous)
            const float* __restrict__ b4, const float* __restrict__ b5,
            const float* __restrict__ sums, const float* __restrict__ cnt,
            float* __restrict__ out) {
  extern __shared__ char smem[];
  _Float16* sW4 = (_Float16*)(smem + N_W4);
  _Float16* sW5 = (_Float16*)(smem + N_W5);
  float*    sB4 = (float*)(smem + N_B4);
  float*    sB5 = (float*)(smem + N_B5);

  const int tid = threadIdx.x;
  // async-stage W4/W5 fragments (51200 B) into LDS
  stage_to_lds(W4p, smem + N_W4, 46080 + 5120, tid, blockDim.x);
  if (tid < 144) sB4[tid] = (tid < 131) ? b4[tid] : 0.f;
  if (tid < 16)  sB5[tid] = (tid < 2) ? b5[tid] : 0.f;
  stage_fence();

  const int lane = tid & 31;
  const int wid  = tid >> 5;
  const int hl   = lane >> 4;
  const int lm   = lane & 15;
  const int nWaves = (gridDim.x * blockDim.x) >> 5;
  const int wId    = (blockIdx.x * blockDim.x + tid) >> 5;
  const int ntiles = NNODES / 16;
  _Float16* my = (_Float16*)(smem + N_SCR) + wid * 2560;   // 16x160 halfs

  for (int t = wId; t < ntiles; t += nWaves) {
    // zero the row scratch (covers K padding 131..159)
    for (int i = lane; i < 16 * 160; i += 32) my[i] = (_Float16)0.f;

    float invL = 0.f;
    if (lane < 16) {
      int n0 = t * 16 + lane;
      invL = 1.0f / fmaxf(cnt[n0], 1.0f);
      my[lane * 160 + 0]   = (_Float16)x[n0 * 2 + 0];
      my[lane * 160 + 1]   = (_Float16)x[n0 * 2 + 1];
      my[lane * 160 + 130] = (_Float16)u[batch[n0]];
    }
    // fill agg = sums * inv (cols 2..129)
    for (int i = lane; i < 16 * 128; i += 32) {
      int row = i >> 7, k = i & 127;
      float iv = __shfl(invL, row, 32);
      my[row * 160 + 2 + k] = (_Float16)(sums[(t * 16 + row) * 128 + k] * iv);
    }

    // A fragments for layer 4 (K = 160 -> 5 chunks)
    v16h a4[5];
#pragma unroll
    for (int kt = 0; kt < 5; ++kt) {
#pragma unroll
      for (int g = 0; g < 2; ++g) {
        int Kb = kt * 32 + g * 16 + (hl << 3);
        v8h c8 = *(const v8h*)&my[lm * 160 + Kb];
#pragma unroll
        for (int q = 0; q < 8; ++q) a4[kt][g * 8 + q] = c8[q];
      }
    }

    // ---- layer 4: h4 = relu(o @ W4 + b4), N padded to 144 (9x5 WMMAs) ----
#pragma unroll
    for (int nt = 0; nt < 9; ++nt) {
      v8f acc = {};
#pragma unroll
      for (int kt = 0; kt < 5; ++kt) {
        v16h b = *(const v16h*)&sW4[frag_idx(kt, nt, 9, lane)];
        acc = __builtin_amdgcn_wmma_f32_16x16x32_f16(false, a4[kt], false, b,
                                                     (short)0, acc, false, false);
      }
      float bias = sB4[nt * 16 + lm];
#pragma unroll
      for (int r = 0; r < 8; ++r) {
        int M = (hl << 3) + r;
        my[M * 160 + nt * 16 + lm] = (_Float16)fmaxf(acc[r] + bias, 0.f);
      }
    }

    // reload h4 as A fragments
    v16h a5[5];
#pragma unroll
    for (int kt = 0; kt < 5; ++kt) {
#pragma unroll
      for (int g = 0; g < 2; ++g) {
        int Kb = kt * 32 + g * 16 + (hl << 3);
        v8h c8 = *(const v8h*)&my[lm * 160 + Kb];
#pragma unroll
        for (int q = 0; q < 8; ++q) a5[kt][g * 8 + q] = c8[q];
      }
    }

    // ---- layer 5: out = h4 @ W5 + b5 (N=2 in one padded tile, 5 WMMAs) ----
    v8f acc = {};
#pragma unroll
    for (int kt = 0; kt < 5; ++kt) {
      v16h b = *(const v16h*)&sW5[frag_idx(kt, 0, 1, lane)];
      acc = __builtin_amdgcn_wmma_f32_16x16x32_f16(false, a5[kt], false, b,
                                                   (short)0, acc, false, false);
    }
    if (lm < 2) {
      float bias = sB5[lm];
#pragma unroll
      for (int r = 0; r < 8; ++r) {
        int node = t * 16 + (hl << 3) + r;
        out[node * 2 + lm] = acc[r] + bias;
      }
    }
  }
}

// ---------------- host launcher ----------------
extern "C" void kernel_launch(void* const* d_in, const int* in_sizes, int n_in,
                              void* d_out, int out_size, void* d_ws, size_t ws_size,
                              hipStream_t stream) {
  (void)in_sizes; (void)n_in; (void)out_size; (void)ws_size;
  const float* x   = (const float*)d_in[0];
  const int*   ei  = (const int*)d_in[1];
  const float* ea  = (const float*)d_in[2];
  const float* u   = (const float*)d_in[3];
  const int*   bat = (const int*)d_in[4];
  const float* W1  = (const float*)d_in[5];
  const float* b1  = (const float*)d_in[6];
  const float* W2  = (const float*)d_in[7];
  const float* b2  = (const float*)d_in[8];
  const float* W3  = (const float*)d_in[9];
  const float* b3  = (const float*)d_in[10];
  const float* W4  = (const float*)d_in[11];
  const float* b4  = (const float*)d_in[12];
  const float* W5  = (const float*)d_in[13];
  const float* b5  = (const float*)d_in[14];
  float* out = (float*)d_out;

  char* ws = (char*)d_ws;
  _Float16* W2p = (_Float16*)(ws + OFF_W2P);
  _Float16* W3p = (_Float16*)(ws + OFF_W3P);
  _Float16* W4p = (_Float16*)(ws + OFF_W4P);
  _Float16* W5p = (_Float16*)(ws + OFF_W5P);
  float* sums = (float*)(ws + OFF_SUMS);
  float* cnt  = (float*)(ws + OFF_CNT);

  // zero accumulators (sums + cnt are contiguous)
  (void)hipMemsetAsync(sums, 0, ((size_t)NNODES * HID + NNODES) * sizeof(float), stream);

  // pre-swizzle weights into f16 WMMA B-fragment layout
  prep_kernel<<<228, 256, 0, stream>>>(W2, W3, W4, W5, W2p, W3p, W4p, W5p);

  // fused edge MLP + scatter-add (one wave per 16-edge tile, grid-strided)
  edge_kernel<<<1024, 256, E_SMEM, stream>>>(x, ei, ea, W1, b1, W2p, b2, b3,
                                             sums, cnt);

  // fused mean + node MLP
  node_kernel<<<512, 256, N_SMEM, stream>>>(x, u, bat, W4p, b4, b5,
                                            sums, cnt, out);
}
